// GEP_63015760167113
// MI455X (gfx1250) — matrix operand order repaired and checked by reading, not compiled, not run on hardware
//
#include <hip/hip_runtime.h>
#include <hip/hip_bf16.h>

typedef __attribute__((ext_vector_type(16))) __bf16 v16bf;
typedef __attribute__((ext_vector_type(8)))  float  v8f;

#define TOTAL_P   917504
#define ESTRIDE   1056      // 480 + 320 + 256 (per-group K padded to mult of 32)
#define EMB_COLS  999       // 453 + 320 + 226
#define BATCH     256
#define KCHUNK    8192

// fp32 -> bf16 bits with round-to-nearest-even
static __device__ __forceinline__ unsigned bfbits(float x) {
    union { float f; unsigned u; } c; c.f = x;
    return (c.u + 0x7fffu + ((c.u >> 16) & 1u)) >> 16;
}
// pack two floats into one dword of 2xbf16 (lo = first/even K)
static __device__ __forceinline__ unsigned packbf(float lo, float hi) {
    return bfbits(lo) | (bfbits(hi) << 16);
}

union fragU { v16bf v; unsigned u[8]; };

// ---------------- Kernel 0: zero workspace accumulators ----------------
__global__ void zero_ws_kernel(float* ws, int n) {
    int i = blockIdx.x * blockDim.x + threadIdx.x;
    if (i < n) ws[i] = 0.0f;
}

// ---------------- Kernel 1: projection GEMM  E += G_chunk @ W_chunk ----------------
// grid: (ceil(ntiles/8), 16, P/KCHUNK), block: 256 (8 waves, one 16-wide N tile each).
// G pre-offset to the group's first column; row stride TOTAL_P. NB compile-time.
template<int NB, int GOFF>
__global__ __launch_bounds__(256)
void proj_kernel(const float* __restrict__ G, const float* __restrict__ W,
                 float* __restrict__ E) {
    __shared__ unsigned gtile[16 * 35];   // 16 rows x 32 dwords (64 K as bf16 pairs), stride 35
    const int tid  = threadIdx.x;
    const int lane = tid & 31;
    const int wv   = tid >> 5;
    const int h    = lane >> 4;
    const int nl   = lane & 15;
    const int m0   = blockIdx.y * 16;
    const int ncol = (blockIdx.x * 8 + wv) * 16 + nl;
    const bool nvalid = (ncol < NB);
    const int  ncc  = nvalid ? ncol : (NB - 1);       // clamped, always in-bounds
    const long k0   = (long)blockIdx.z * KCHUNK;

    // staging: 256 threads, each 4 consecutive floats (one float4) of the 16x64 tile
    const int grow = tid >> 4;
    const int gc4  = (tid & 15) * 4;
    const float* gbase = G + (long)(m0 + grow) * TOTAL_P + gc4;
    unsigned* gd = &gtile[grow * 35 + (tid & 15) * 2];

    v8f acc = {};
    for (int kb = 0; kb < KCHUNK; kb += 64) {
        const long k = k0 + kb;
        float4 a0 = *(const float4*)(gbase + k);
        if (kb + 64 < KCHUNK)
            __builtin_prefetch(gbase + k + 512, 0, 3);
        gd[0] = packbf(a0.x, a0.y);
        gd[1] = packbf(a0.z, a0.w);
        __syncthreads();

        #pragma unroll
        for (int s = 0; s < 2; ++s) {
            fragU A;                                   // 16x32 of G (bf16 pairs from LDS)
            #pragma unroll
            for (int j = 0; j < 8; ++j)
                A.u[j] = gtile[nl * 35 + s * 16 + 8 * h + j];
            fragU B;                                   // 32x16 of W, immediate-offset loads
            const float* wp = W + (long)(k + s * 32 + h * 16) * NB + ncc;
            #pragma unroll
            for (int j = 0; j < 8; ++j) {
                float v0 = wp[(2 * j) * NB];
                float v1 = wp[(2 * j + 1) * NB];
                if (!nvalid) { v0 = 0.0f; v1 = 0.0f; } // branchless select
                B.u[j] = packbf(v0, v1);
            }
            acc = __builtin_amdgcn_wmma_f32_16x16x32_bf16(false, A.v, false, B.v,
                                                          (short)0, acc, false, false);
        }
        __syncthreads();
    }
    if (nvalid) {
        #pragma unroll
        for (int r = 0; r < 8; ++r) {
            const int brow = m0 + r + 8 * h;          // C/D layout: M = r + 8h, N = lane&15
            atomicAdd(&E[(long)brow * ESTRIDE + GOFF + ncol], acc[r]);
        }
    }
}

// ---------------- Kernel 2: clip embedding rows, write packed output ----------------
__global__ __launch_bounds__(256)
void embclip_kernel(const float* __restrict__ E, float* __restrict__ out) {
    __shared__ float red[256];
    const int b = blockIdx.x, t = threadIdx.x;
    float ss = 0.0f;
    for (int j = t; j < ESTRIDE; j += 256) {          // padded cols are zero
        const float v = E[(long)b * ESTRIDE + j];
        ss += v * v;
    }
    red[t] = ss; __syncthreads();
    for (int s = 128; s > 0; s >>= 1) {
        if (t < s) red[t] += red[t + s];
        __syncthreads();
    }
    const float scale = fminf(1.0f / sqrtf(red[0]), 1.0f);
    for (int j = t; j < EMB_COLS; j += 256) {
        const int jp = (j < 453) ? j : j + 27;        // packed -> padded column map
        out[(long)b * EMB_COLS + j] = E[(long)b * ESTRIDE + jp] * scale;
    }
}

// ---------------- Kernel 3: residual R = G - E @ W^T, accumulate row sumsq ----------------
// grid: P/128, block: 256 (8 waves; wave w owns N tile w, all 256 batch rows)
template<int NB, int NSTEPS, int GOFF, long POFF>
__global__ __launch_bounds__(256)
void resid_kernel(const float* __restrict__ G, const float* __restrict__ W,
                  const float* __restrict__ E, float* __restrict__ Rout,
                  float* __restrict__ rsglob) {
    __shared__ unsigned wtile[128 * 19];  // 128 p-rows x 16 dwords (32 K bf16), stride 19
    __shared__ unsigned etile[256 * 19];  // 256 b-rows x 16 dwords (32 K bf16), stride 19
    __shared__ float rsl[256];
    const int tid  = threadIdx.x;
    const int lane = tid & 31;
    const int wv   = tid >> 5;
    const int h    = lane >> 4;
    const int nl   = lane & 15;
    const long p0  = (long)blockIdx.x * 128;
    rsl[tid] = 0.0f;

    v8f acc[16];
    #pragma unroll
    for (int m = 0; m < 16; ++m) acc[m] = (v8f){};

    for (int ks = 0; ks < NSTEPS; ++ks) {
        const int kbase = ks * 32;
        // stage W^T tile (convert to bf16 pairs): 128x16 dwords, 8 per thread
        #pragma unroll
        for (int i = 0; i < 8; ++i) {
            const int idx = tid + i * 256;
            const int r = idx >> 4, c = idx & 15, kk = kbase + 2 * c;
            const float v0 = (kk     < NB) ? W[(p0 + r) * (long)NB + kk]     : 0.0f;
            const float v1 = (kk + 1 < NB) ? W[(p0 + r) * (long)NB + kk + 1] : 0.0f;
            wtile[r * 19 + c] = packbf(v0, v1);
        }
        // stage E tile: 256x16 dwords, 16 per thread (padded zeros already in E)
        #pragma unroll
        for (int i = 0; i < 16; ++i) {
            const int idx = tid + i * 256;
            const int r = idx >> 4, c = idx & 15;
            const float2 ev = *(const float2*)&E[(long)r * ESTRIDE + GOFF + kbase + 2 * c];
            etile[r * 19 + c] = packbf(ev.x, ev.y);
        }
        __syncthreads();

        fragU B;                                      // B[k][n] = W[p0 + wv*16 + n][k]
        #pragma unroll
        for (int j = 0; j < 8; ++j)
            B.u[j] = wtile[(wv * 16 + nl) * 19 + h * 8 + j];

        #pragma unroll
        for (int m = 0; m < 16; ++m) {
            fragU A;
            #pragma unroll
            for (int j = 0; j < 8; ++j)
                A.u[j] = etile[(m * 16 + nl) * 19 + h * 8 + j];
            acc[m] = __builtin_amdgcn_wmma_f32_16x16x32_bf16(false, A.v, false, B.v,
                                                             (short)0, acc[m], false, false);
        }
        __syncthreads();
    }

    const long pcol = p0 + wv * 16 + nl;
    #pragma unroll
    for (int m = 0; m < 16; ++m) {
        #pragma unroll
        for (int r = 0; r < 8; ++r) {
            const int brow = m * 16 + r + 8 * h;
            const long gidx = (long)brow * TOTAL_P + POFF + pcol;
            const float v = G[gidx] - acc[m][r];
            Rout[gidx] = v;
            float s = v * v;                          // reduce over 16 N-lanes (same row)
            s += __shfl_xor(s, 1); s += __shfl_xor(s, 2);
            s += __shfl_xor(s, 4); s += __shfl_xor(s, 8);
            if (nl == 0) atomicAdd(&rsl[brow], s);
        }
    }
    __syncthreads();
    atomicAdd(&rsglob[tid], rsl[tid]);
}

// ---------------- Kernel 4: scale residual rows by min(1/norm, 1) ----------------
__global__ __launch_bounds__(256)
void scale_resid_kernel(float* __restrict__ R, const float* __restrict__ rs, long n4) {
    const long i = (long)blockIdx.x * blockDim.x + threadIdx.x;
    if (i >= n4) return;
    const long e0 = i * 4;
    const int row = (int)(e0 / TOTAL_P);
    const float scale = fminf(1.0f / sqrtf(rs[row]), 1.0f);
    float4 v = ((float4*)R)[i];
    v.x *= scale; v.y *= scale; v.z *= scale; v.w *= scale;
    ((float4*)R)[i] = v;
}

extern "C" void kernel_launch(void* const* d_in, const int* in_sizes, int n_in,
                              void* d_out, int out_size, void* d_ws, size_t ws_size,
                              hipStream_t stream) {
    const float* tg = (const float*)d_in[0];
    const float* W0 = (const float*)d_in[1];
    const float* W1 = (const float*)d_in[2];
    const float* W2 = (const float*)d_in[3];

    float* E  = (float*)d_ws;                   // [256 x 1056] f32
    float* rs = E + (long)BATCH * ESTRIDE;      // [256] f32 residual row sumsq
    float* out_emb = (float*)d_out;                         // [256 x 999]
    float* out_res = out_emb + (long)BATCH * EMB_COLS;      // [256 x 917504]

    // 0) zero accumulators
    const int nzero = BATCH * ESTRIDE + BATCH;
    zero_ws_kernel<<<(nzero + 255) / 256, 256, 0, stream>>>(E, nzero);

    // 1) projection GEMMs (split-K, bf16 WMMA, f32 atomic reduction)
    {   // group 0: P=524288, NB=453 -> 29 n-tiles -> grid.x = 4
        dim3 g(4, 16, 524288 / KCHUNK);
        proj_kernel<453, 0><<<g, 256, 0, stream>>>(tg, W0, E);
    }
    {   // group 1: P=262144, NB=320 -> 20 n-tiles -> grid.x = 3
        dim3 g(3, 16, 262144 / KCHUNK);
        proj_kernel<320, 480><<<g, 256, 0, stream>>>(tg + 524288, W1, E);
    }
    {   // group 2: P=131072, NB=226 -> 15 n-tiles -> grid.x = 2
        dim3 g(2, 16, 131072 / KCHUNK);
        proj_kernel<226, 800><<<g, 256, 0, stream>>>(tg + 786432, W2, E);
    }

    // 2) embedding row clip -> output
    embclip_kernel<<<BATCH, 256, 0, stream>>>(E, out_emb);

    // 3) residual GEMMs (unclipped E) + row sumsq accumulation
    resid_kernel<453, 15, 0,   0L     ><<<524288 / 128, 256, 0, stream>>>(tg, W0, E, out_res, rs);
    resid_kernel<320, 10, 480, 524288L><<<262144 / 128, 256, 0, stream>>>(tg, W1, E, out_res, rs);
    resid_kernel<226, 8,  800, 786432L><<<131072 / 128, 256, 0, stream>>>(tg, W2, E, out_res, rs);

    // 4) residual row clip (in place, float4)
    const long n4 = (long)BATCH * TOTAL_P / 4;
    scale_resid_kernel<<<(unsigned)((n4 + 255) / 256), 256, 0, stream>>>(out_res, rs, n4);
}
